// BaseMultiHeadAttention_29807073034579
// MI455X (gfx1250) — compile-verified, matrix operand order checked by reading, not run on hardware
//
#include <hip/hip_runtime.h>

// ---------------------------------------------------------------------------
// MI455X (gfx1250) fused MHA: RMSNorm+RoPE -> flash attention -> projection.
// Matrix math:   v_wmma_f32_16x16x32_f16 (wave32, f32 accumulate).
// Data movement: Tensor Data Mover (tensor_load_to_lds) double-buffered K/V/W
//                tiles with TDM row padding for bank-conflict-free LDS reads,
//                completion via TENSORcnt (s_wait_tensorcnt) + block barrier.
// Fragments for a WMMA group are loaded into distinct registers before the
// WMMAs so LDS latency overlaps matrix issue (partial s_wait_dscnt).
// Workspace layout (assumes ws_size >= 66 MB):
//   [ 0,16MB) qh   f16 [B,H,S,64]
//   [16,32MB) kh   f16 [B,H,S,64]
//   [32,48MB) vT   f16 [B,H,64,S]      (V transposed; TDM gathers 2-D tiles)
//   [48,64MB) c16  f16 [B,S,E]         (attention output, GEMM A operand)
//   [64,66MB) w16  f16 [E,E]           (proj_w converted once per launch)
// ---------------------------------------------------------------------------

typedef __attribute__((ext_vector_type(16))) _Float16 v16h;
typedef __attribute__((ext_vector_type(8)))  _Float16 v8h;
typedef __attribute__((ext_vector_type(2)))  _Float16 v2h;
typedef __attribute__((ext_vector_type(8)))  float    v8f;
typedef __attribute__((ext_vector_type(4)))  unsigned int u32x4;
typedef __attribute__((ext_vector_type(8)))  int      i32x8;
typedef __attribute__((ext_vector_type(4)))  int      i32x4;

#define S_LEN 2048
#define NHEAD 16
#define HDIM  64
#define EDIM  1024
#define BSZ   4
#define EPS   1.1920928955078125e-07f
// softmax scale * log2(e):  HD^-0.5 * 1.4426950408889634
#define KL2E  (0.125f * 1.4426950408889634f)

__device__ __forceinline__ v8f wmma16(v16h a, v16h b, v8f c) {
  // D = A(16x32 f16) * B(32x16 f16) + C(16x16 f32)
  return __builtin_amdgcn_wmma_f32_16x16x32_f16(
      /*neg_a=*/false, a, /*neg_b=*/false, b,
      /*c_mod=*/(short)0, c, /*reuse_a=*/false, /*reuse_b=*/false);
}

// One lane's A-/B-fragment slice of a 32-wide K-chunk from a row-major row.
// ISA 7.12.2 (16-bit, 16x32): lanes 0-15 hold K={0..7,16..23},
// lanes 16-31 hold K={8..15,24..31}.  hl = lane>>4.
__device__ __forceinline__ v16h ld_frag(const _Float16* __restrict__ row, int hl) {
  const v8h lo = *reinterpret_cast<const v8h*>(row + hl * 8);
  const v8h hi = *reinterpret_cast<const v8h*>(row + hl * 8 + 16);
  v16h r;
#pragma unroll
  for (int i = 0; i < 8; ++i) { r[i] = lo[i]; r[i + 8] = hi[i]; }
  return r;
}

// Flat shared pointer -> LDS byte address (ISA 10.2: low 32 bits of the flat address).
__device__ __forceinline__ unsigned int lds_addr_of(const void* p) {
  return (unsigned int)(size_t)p;
}

// ---------------------------------------------------------------------------
// TDM 2-D tile load: Global -> LDS.  Builds a D# (ISA 8.3/8.4):
//  group0: count=1 | lds_addr | global_addr | type=2
//  group1: data_size=2B, tile_dim0/1, tensor dims, dim0 stride, LDS padding
// pad_int/pad_amt are the *encoded* fields: pad every 2^(pad_int+1) dwords by
// (pad_amt+1) dwords.  6-arg builtin on this toolchain; groups 2/3 unused for
// <=2-D tiles.  Completion tracked with TENSORcnt.
// ---------------------------------------------------------------------------
__device__ __forceinline__ void tdm_load_2d(
    const void* gaddr, unsigned int lds_addr,
    unsigned int tile_d0, unsigned int tile_d1,
    unsigned long long stride0_elems, unsigned int tensor_d0,
    unsigned int pad_int, unsigned int pad_amt) {
  const unsigned long long ga = (unsigned long long)(size_t)gaddr;
  u32x4 g0;
  g0[0] = 1u;                                              // count=1
  g0[1] = lds_addr;                                        // LDS byte addr
  g0[2] = (unsigned int)ga;                                // global_addr lo
  g0[3] = (unsigned int)((ga >> 32) & 0x1FFFFFFu) | (2u << 30);  // hi | type=2
  i32x8 g1;
  g1[0] = (int)((1u << 16)                                 // data_size = 2B
                | (1u << 20)                               // pad_enable
                | (pad_int << 22) | (pad_amt << 25));
  g1[1] = (int)((tensor_d0 & 0xFFFFu) << 16);              // tensor_dim0 lo
  g1[2] = (int)(((tensor_d0 >> 16) & 0xFFFFu)              // tensor_dim0 hi
                | (0x0000u << 16));                        // tensor_dim1 lo
  g1[3] = (int)((0x0010u)                                  // tensor_dim1 hi (big)
                | (tile_d0 << 16));                        // tile_dim0
  g1[4] = (int)(tile_d1 & 0xFFFFu);                        // tile_dim1 (dim2=0)
  g1[5] = (int)(unsigned int)(stride0_elems & 0xFFFFFFFFu);// dim0 stride lo
  g1[6] = (int)(unsigned int)((stride0_elems >> 32) & 0xFFFFu); // stride hi
  g1[7] = 0;
  const i32x4 z4 = {0, 0, 0, 0};
  const i32x8 z8 = {0, 0, 0, 0, 0, 0, 0, 0};
  __builtin_amdgcn_tensor_load_to_lds(g0, g1, z4, z4, z8, 0);
}

// ---------------------------------------------------------------------------
// Kernel 1: per-(b,s,h) RMSNorm + RoPE for Q,K (f16 out, head-major) and
// V convert+transpose. One wave owns one 64-elem head vector (1 RoPE pair/lane).
// ---------------------------------------------------------------------------
__global__ __launch_bounds__(256) void prep_qkv(
    const float* __restrict__ q, const float* __restrict__ k,
    const float* __restrict__ v, const float* __restrict__ qw,
    const float* __restrict__ kw,
    _Float16* __restrict__ qh, _Float16* __restrict__ kh,
    _Float16* __restrict__ vT) {
  const int lane = threadIdx.x & 31;
  const int idx  = blockIdx.x * 8 + (threadIdx.x >> 5);  // (b,s,h) flat
  const int h = idx & (NHEAD - 1);
  const int s = (idx >> 4) & (S_LEN - 1);
  const int b = idx >> 15;

  const size_t in_off  = ((size_t)(b * S_LEN + s)) * EDIM + h * HDIM + 2 * lane;
  const size_t out_off = (((size_t)(b * NHEAD + h)) * S_LEN + s) * HDIM + 2 * lane;

  const float inv_freq = exp2f(-(float)lane * 0.41524101186092026f); // (2/64)*log2(1e4)
  float sn, cs;
  __sincosf((float)s * inv_freq, &sn, &cs);

  {  // Q
    float x1 = q[in_off], x2 = q[in_off + 1];
    float ss = x1 * x1 + x2 * x2;
    ss += __shfl_xor(ss, 1);  ss += __shfl_xor(ss, 2);
    ss += __shfl_xor(ss, 4);  ss += __shfl_xor(ss, 8);
    ss += __shfl_xor(ss, 16);
    const float r = rsqrtf(ss * (1.0f / HDIM) + EPS);
    const float n1 = x1 * r * qw[2 * lane], n2 = x2 * r * qw[2 * lane + 1];
    v2h o; o[0] = (_Float16)(n1 * cs - n2 * sn); o[1] = (_Float16)(n1 * sn + n2 * cs);
    *reinterpret_cast<v2h*>(qh + out_off) = o;
  }
  {  // K
    float x1 = k[in_off], x2 = k[in_off + 1];
    float ss = x1 * x1 + x2 * x2;
    ss += __shfl_xor(ss, 1);  ss += __shfl_xor(ss, 2);
    ss += __shfl_xor(ss, 4);  ss += __shfl_xor(ss, 8);
    ss += __shfl_xor(ss, 16);
    const float r = rsqrtf(ss * (1.0f / HDIM) + EPS);
    const float n1 = x1 * r * kw[2 * lane], n2 = x2 * r * kw[2 * lane + 1];
    v2h o; o[0] = (_Float16)(n1 * cs - n2 * sn); o[1] = (_Float16)(n1 * sn + n2 * cs);
    *reinterpret_cast<v2h*>(kh + out_off) = o;
  }
  {  // V -> [B,H,HD,S]
    const float x1 = v[in_off], x2 = v[in_off + 1];
    const size_t t0 = (((size_t)(b * NHEAD + h)) * HDIM + 2 * lane) * S_LEN + s;
    vT[t0]         = (_Float16)x1;
    vT[t0 + S_LEN] = (_Float16)x2;
  }
}

// ---------------------------------------------------------------------------
// Kernel 2: proj_w f32 -> f16
// ---------------------------------------------------------------------------
__global__ __launch_bounds__(256) void conv_w(const float* __restrict__ w,
                                              _Float16* __restrict__ w16) {
  const int i = blockIdx.x * 256 + threadIdx.x;
  w16[i] = (_Float16)w[i];
}

// ---------------------------------------------------------------------------
// Kernel 3: flash attention. Block = 4 waves owning (b,h,64 q-rows); wave =
// one 16-row Q tile.  K/V chunks staged by TDM, double buffered:
//   K tile: 32 keys x 64d, padded to 144B rows (16 distinct banks for B128)
//   V tile: 64d x 32 keys from vT, padded to 80B rows
// Per 32-key chunk: 4 WMMAs scores, online softmax (16-lane shfl reductions in
// WMMA C layout), P via padded LDS (C->A layout), 4 WMMAs PV.
// ---------------------------------------------------------------------------
__global__ __launch_bounds__(128) void flash_attn(
    const _Float16* __restrict__ qh, const _Float16* __restrict__ kh,
    const _Float16* __restrict__ vT, _Float16* __restrict__ c16) {
  __shared__ _Float16 kT[2][32 * 72];   // 32 key rows, stride 72 halfs (144B)
  __shared__ _Float16 vTl[2][64 * 40];  // 64 d rows,  stride 40 halfs (80B)
  __shared__ _Float16 pTile[4][16 * 40];

  const int lane = threadIdx.x & 31;
  const int wv   = threadIdx.x >> 5;
  const int l16  = lane & 15;
  const int hl   = lane >> 4;

  const int qblk = blockIdx.x & 31;
  const int h    = (blockIdx.x >> 5) & (NHEAD - 1);
  const int b    = blockIdx.x >> 9;
  const int q0   = qblk * 64 + wv * 16;

  const _Float16* qb = qh + ((size_t)(b * NHEAD + h)) * S_LEN * HDIM;
  const _Float16* kb = kh + ((size_t)(b * NHEAD + h)) * S_LEN * HDIM;
  const _Float16* vb = vT + ((size_t)(b * NHEAD + h)) * HDIM * S_LEN;

  // Q A-fragments for HD chunks [0,32) and [32,64); live in VGPRs all loop
  const v16h qf0 = ld_frag(qb + (size_t)(q0 + l16) * HDIM + 0,  hl);
  const v16h qf1 = ld_frag(qb + (size_t)(q0 + l16) * HDIM + 32, hl);

  // TDM prologue: stage chunk 0 into buffer 0 (wave 0 drives the DMA)
  if (wv == 0) {
    tdm_load_2d(kb, lds_addr_of(&kT[0][0]), /*t0=*/64, /*t1=*/32,
                /*stride0=*/HDIM, /*tdim0=*/HDIM, /*pad: 32dw->*/4, /*4dw*/3);
    tdm_load_2d(vb, lds_addr_of(&vTl[0][0]), /*t0=*/32, /*t1=*/64,
                /*stride0=*/S_LEN, /*tdim0=*/S_LEN, /*pad: 16dw->*/3, /*4dw*/3);
  }

  v8f acc[4];
  float m[8], l[8];
#pragma unroll
  for (int t = 0; t < 4; ++t)
#pragma unroll
    for (int r = 0; r < 8; ++r) acc[t][r] = 0.0f;
#pragma unroll
  for (int r = 0; r < 8; ++r) { m[r] = -1e30f; l[r] = 0.0f; }

  _Float16* pw = pTile[wv];

  int buf = 0;
  for (int kc = 0; kc < S_LEN; kc += 32, buf ^= 1) {
    // ---- prefetch next chunk, then wait for current chunk's 2 TDM ops ----
    if (wv == 0) {
      if (kc + 32 < S_LEN) {
        tdm_load_2d(kb + (size_t)(kc + 32) * HDIM, lds_addr_of(&kT[buf ^ 1][0]),
                    64, 32, HDIM, HDIM, 4, 3);
        tdm_load_2d(vb + (kc + 32), lds_addr_of(&vTl[buf ^ 1][0]),
                    32, 64, S_LEN, S_LEN, 3, 3);
        __builtin_amdgcn_s_wait_tensorcnt(2);
      } else {
        __builtin_amdgcn_s_wait_tensorcnt(0);
      }
    }
    __syncthreads();  // current-buffer tiles visible to all 4 waves

    const _Float16* kTb = kT[buf];
    const _Float16* vTb = vTl[buf];

    // ---- scores: 16 rows x 32 keys.  Load all 4 K fragments into distinct
    // registers first so LDS latency overlaps WMMA issue. ----
    const v16h kf00 = ld_frag(kTb + (l16) * 72 + 0,       hl);
    const v16h kf01 = ld_frag(kTb + (l16) * 72 + 32,      hl);
    const v16h kf10 = ld_frag(kTb + (16 + l16) * 72 + 0,  hl);
    const v16h kf11 = ld_frag(kTb + (16 + l16) * 72 + 32, hl);
    v8f sc0, sc1;
#pragma unroll
    for (int r = 0; r < 8; ++r) { sc0[r] = 0.0f; sc1[r] = 0.0f; }
    sc0 = wmma16(qf0, kf00, sc0);
    sc1 = wmma16(qf0, kf10, sc1);
    sc0 = wmma16(qf1, kf01, sc0);
    sc1 = wmma16(qf1, kf11, sc1);

    // ---- online softmax (C layout: row r+8*hl, col l16) ----
    float fac[8];
#pragma unroll
    for (int r = 0; r < 8; ++r) {
      float mx = fmaxf(sc0[r], sc1[r]);
      mx = fmaxf(mx, __shfl_xor(mx, 1));
      mx = fmaxf(mx, __shfl_xor(mx, 2));
      mx = fmaxf(mx, __shfl_xor(mx, 4));
      mx = fmaxf(mx, __shfl_xor(mx, 8));
      const float mn = fmaxf(m[r], mx);
      fac[r] = exp2f((m[r] - mn) * KL2E);
      m[r] = mn;
    }
#pragma unroll
    for (int r = 0; r < 8; ++r) {
      const float p0 = exp2f((sc0[r] - m[r]) * KL2E);
      const float p1 = exp2f((sc1[r] - m[r]) * KL2E);
      pw[(r + 8 * hl) * 40 + l16]      = (_Float16)p0;
      pw[(r + 8 * hl) * 40 + 16 + l16] = (_Float16)p1;
      float s = p0 + p1;
      s += __shfl_xor(s, 1); s += __shfl_xor(s, 2);
      s += __shfl_xor(s, 4); s += __shfl_xor(s, 8);
      l[r] = l[r] * fac[r] + s;
#pragma unroll
      for (int t = 0; t < 4; ++t) acc[t][r] *= fac[r];
    }
    __syncthreads();  // LDS P writes ordered before A-fragment reads

    // ---- P (A layout) + all 4 V fragments, then 4 WMMAs ----
    v16h pf;
    {
      const v8h lo = *reinterpret_cast<const v8h*>(&pw[l16 * 40 + hl * 8]);
      const v8h hi = *reinterpret_cast<const v8h*>(&pw[l16 * 40 + 16 + hl * 8]);
#pragma unroll
      for (int i = 0; i < 8; ++i) { pf[i] = lo[i]; pf[i + 8] = hi[i]; }
    }
    const v16h vf0 = ld_frag(vTb + (0 * 16 + l16) * 40, hl);
    const v16h vf1 = ld_frag(vTb + (1 * 16 + l16) * 40, hl);
    const v16h vf2 = ld_frag(vTb + (2 * 16 + l16) * 40, hl);
    const v16h vf3 = ld_frag(vTb + (3 * 16 + l16) * 40, hl);
    acc[0] = wmma16(pf, vf0, acc[0]);
    acc[1] = wmma16(pf, vf1, acc[1]);
    acc[2] = wmma16(pf, vf2, acc[2]);
    acc[3] = wmma16(pf, vf3, acc[3]);
    __syncthreads();  // all waves done with buf before TDM refills it
  }

  // ---- normalize and store ctx (f16) into [B,S,E] at cols h*64+.. ----
#pragma unroll
  for (int r = 0; r < 8; ++r) {
    const float inv = 1.0f / l[r];
    const size_t row = (size_t)b * S_LEN + (q0 + r + 8 * hl);
#pragma unroll
    for (int t = 0; t < 4; ++t)
      c16[row * EDIM + h * HDIM + t * 16 + l16] = (_Float16)(acc[t][r] * inv);
  }
}

// ---------------------------------------------------------------------------
// Kernel 4: out[M=8192,N=1024] = c16 @ w16^T + bias, f32 out.
// Block = 4 waves, 64x64 tile; W tiles (shared by all 4 waves) staged by TDM
// double-buffered with padded rows; A rows (wave-private) direct from global.
// ---------------------------------------------------------------------------
__global__ __launch_bounds__(128) void proj_gemm(
    const _Float16* __restrict__ c16, const _Float16* __restrict__ w16,
    const float* __restrict__ bias, float* __restrict__ out) {
  __shared__ _Float16 wT[2][64 * 40];  // 64 rows of W, stride 40 halfs (80B)

  const int lane = threadIdx.x & 31;
  const int wv   = threadIdx.x >> 5;
  const int l16  = lane & 15;
  const int hl   = lane >> 4;
  const int nblk = blockIdx.x & 15;
  const int mblk = blockIdx.x >> 4;
  const int m0 = mblk * 64 + wv * 16;
  const int n0 = nblk * 64;

  if (wv == 0)
    tdm_load_2d(w16 + (size_t)n0 * EDIM, lds_addr_of(&wT[0][0]),
                /*t0=*/32, /*t1=*/64, /*stride0=*/EDIM, EDIM, 3, 3);

  v8f acc[4];
#pragma unroll
  for (int t = 0; t < 4; ++t)
#pragma unroll
    for (int r = 0; r < 8; ++r) acc[t][r] = 0.0f;

  int buf = 0;
  for (int kc = 0; kc < EDIM; kc += 32, buf ^= 1) {
    if (wv == 0) {
      if (kc + 32 < EDIM) {
        tdm_load_2d(w16 + (size_t)n0 * EDIM + kc + 32, lds_addr_of(&wT[buf ^ 1][0]),
                    32, 64, EDIM, EDIM, 3, 3);
        __builtin_amdgcn_s_wait_tensorcnt(1);
      } else {
        __builtin_amdgcn_s_wait_tensorcnt(0);
      }
    }
    __syncthreads();

    const _Float16* wTb = wT[buf];
    const v16h a   = ld_frag(c16 + (size_t)(m0 + l16) * EDIM + kc, hl);
    const v16h bf0 = ld_frag(wTb + (0 * 16 + l16) * 40, hl);
    const v16h bf1 = ld_frag(wTb + (1 * 16 + l16) * 40, hl);
    const v16h bf2 = ld_frag(wTb + (2 * 16 + l16) * 40, hl);
    const v16h bf3 = ld_frag(wTb + (3 * 16 + l16) * 40, hl);
    acc[0] = wmma16(a, bf0, acc[0]);
    acc[1] = wmma16(a, bf1, acc[1]);
    acc[2] = wmma16(a, bf2, acc[2]);
    acc[3] = wmma16(a, bf3, acc[3]);
    __syncthreads();
  }
#pragma unroll
  for (int t = 0; t < 4; ++t) {
    const float bv = bias[n0 + t * 16 + l16];
#pragma unroll
    for (int r = 0; r < 8; ++r)
      out[(size_t)(m0 + r + 8 * hl) * EDIM + n0 + t * 16 + l16] = acc[t][r] + bv;
  }
}

// ---------------------------------------------------------------------------
extern "C" void kernel_launch(void* const* d_in, const int* in_sizes, int n_in,
                              void* d_out, int out_size, void* d_ws, size_t ws_size,
                              hipStream_t stream) {
  const float* q  = (const float*)d_in[0];
  const float* k  = (const float*)d_in[1];
  const float* v  = (const float*)d_in[2];
  const float* qw = (const float*)d_in[3];
  const float* kw = (const float*)d_in[4];
  const float* pw = (const float*)d_in[5];
  const float* pb = (const float*)d_in[6];

  char* ws = (char*)d_ws;
  const size_t MB = 1024ull * 1024ull;
  _Float16* qh  = (_Float16*)(ws + 0 * MB);
  _Float16* kh  = (_Float16*)(ws + 16 * MB);
  _Float16* vT  = (_Float16*)(ws + 32 * MB);
  _Float16* c16 = (_Float16*)(ws + 48 * MB);
  _Float16* w16 = (_Float16*)(ws + 64 * MB);

  prep_qkv<<<(BSZ * S_LEN * NHEAD) / 8, 256, 0, stream>>>(q, k, v, qw, kw, qh, kh, vT);
  conv_w<<<(EDIM * EDIM) / 256, 256, 0, stream>>>(pw, w16);
  flash_attn<<<BSZ * NHEAD * (S_LEN / 64), 128, 0, stream>>>(qh, kh, vT, c16);
  proj_gemm<<<(BSZ * S_LEN / 64) * (EDIM / 64), 128, 0, stream>>>(c16, w16, pb, (float*)d_out);
}
